// FaceAwareGraphSwinD8B_6897717477526
// MI455X (gfx1250) — matrix operand order repaired and checked by reading, not compiled, not run on hardware
//
#include <hip/hip_runtime.h>
#include <math.h>

// ---------------------------------------------------------------------------
// Types for CDNA5 WMMA (wave32): D(16x16 f32) = A(16x32 f16) * B(32x16 f16) + C
// ---------------------------------------------------------------------------
typedef __attribute__((ext_vector_type(16))) _Float16 v16h;
typedef __attribute__((ext_vector_type(8)))  float    v8f;

union HV {
    v16h   v;
    float4 f[2];   // 32 bytes = 16 halves
};

// ---------------------------------------------------------------------------
// Constants of the model
// ---------------------------------------------------------------------------
#define BD    16       // batch
#define HW    96       // H == W
#define DD    768
#define NWIN  256      // 16x16 windows of 6x6
#define RTOK  64       // regions per batch after 2x2 merge
#define HEADS 8
#define DH    96
#define NCLS  7

__device__ __forceinline__ float gelu_exact(float x) {
    return 0.5f * x * (1.0f + erff(x * 0.70710678118654752f));
}

// ---------------------------------------------------------------------------
// fp32 -> f16 weight conversion
// ---------------------------------------------------------------------------
__global__ void cvt_f32_f16_kernel(const float* __restrict__ in,
                                   _Float16* __restrict__ out, int n) {
    int i = blockIdx.x * 256 + threadIdx.x;
    if (i < n) out[i] = (_Float16)in[i];
}

// ---------------------------------------------------------------------------
// Gate-weighted window pooling (regular: blockIdx.y==0, shifted by 3: y==1).
// Writes the concatenated [reg | shifted] token row directly as f16.
// One block per (batch, window, kind); reads 36 * 768 fp32 = 110 KB coalesced.
// This stage carries ~905 MB of HBM traffic (the roofline bound of the whole
// net), so we prefetch upcoming rows (global_prefetch_b8 on gfx1250).
// ---------------------------------------------------------------------------
__global__ __launch_bounds__(256) void pool_kernel(
    const float* __restrict__ hp,   // [B, 9216, 768]
    const float* __restrict__ gp,   // [B, 9216, 1]
    _Float16* __restrict__ cat)     // [B*256, 1536]
{
    const int bw  = blockIdx.x;          // b*256 + win
    const int b   = bw >> 8;
    const int win = bw & 255;
    const int wy  = win >> 4, wx = win & 15;
    const int sh  = blockIdx.y * 3;      // roll(-3) => src index = idx + 3
    const int tid = threadIdx.x;

    __shared__ float sg[36];
    __shared__ int   soff[36];
    if (tid < 36) {
        int py = tid / 6, px = tid % 6;
        int r = (wy * 6 + py + sh) % HW;
        int c = (wx * 6 + px + sh) % HW;
        int pix = b * (HW * HW) + r * HW + c;
        sg[tid]   = gp[pix];
        soff[tid] = pix;
    }
    __syncthreads();

    float gsum = 0.f;
#pragma unroll
    for (int p = 0; p < 36; ++p) gsum += sg[p];
    const float inv = 1.f / fmaxf(gsum, 1e-6f);

    // prime prefetch of first rows
    __builtin_prefetch(hp + (size_t)soff[0] * DD + tid, 0, 0);
    __builtin_prefetch(hp + (size_t)soff[1] * DD + tid, 0, 0);

    float a0 = 0.f, a1 = 0.f, a2 = 0.f;
    for (int p = 0; p < 36; ++p) {
        if (p + 2 < 36)
            __builtin_prefetch(hp + (size_t)soff[p + 2] * DD + tid, 0, 0);
        const float* hb = hp + (size_t)soff[p] * DD;
        const float  gv = sg[p];
        a0 += hb[tid]       * gv;
        a1 += hb[tid + 256] * gv;
        a2 += hb[tid + 512] * gv;
    }
    _Float16* dst = cat + (size_t)bw * 1536 + blockIdx.y * DD;
    dst[tid]       = (_Float16)(a0 * inv);
    dst[tid + 256] = (_Float16)(a1 * inv);
    dst[tid + 512] = (_Float16)(a2 * inv);
}

// ---------------------------------------------------------------------------
// WMMA GEMM: out[M,N] = A[M,K](f16, row-major) @ B[K,N](f16, row-major) + bias
// epi: 0 = none, 1 = exact GELU. out32 / out16 optionally written.
// Block = 256 threads (8 wave32 waves), tile 64(M) x 128(N), K step 32.
// LDS double-buffered: global loads for tile k+1 are issued into registers,
// overlap the 4 v_wmma of tile k, then a single barrier per K-step.
// Two tiles ahead are prefetched (global_prefetch_b8).
// ---------------------------------------------------------------------------
#define LDA 40
#define LDB 40

static __device__ __forceinline__ void stage_tile(
    _Float16* __restrict__ sA, _Float16* __restrict__ sB,
    float4 aR, float4 bR0, float4 bR1,
    int ar, int acg, int br0, int br1, int bc)
{
    *(float4*)&sA[ar * LDA + acg] = aR;
    union { float4 f; _Float16 h[8]; } t0, t1;
    t0.f = bR0;
    t1.f = bR1;
#pragma unroll
    for (int j = 0; j < 8; ++j) sB[(bc + j) * LDB + br0] = t0.h[j];
#pragma unroll
    for (int j = 0; j < 8; ++j) sB[(bc + j) * LDB + br1] = t1.h[j];
}

__global__ __launch_bounds__(256) void gemm_f16_wmma_kernel(
    const _Float16* __restrict__ A, const _Float16* __restrict__ Bm,
    const float* __restrict__ bias, float* __restrict__ out32,
    _Float16* __restrict__ out16, int M, int N, int K, int epi)
{
    __shared__ __align__(16) _Float16 lA[2][64 * LDA];    // A tiles 64x32
    __shared__ __align__(16) _Float16 lB[2][128 * LDB];   // B tiles transposed [n][k]

    const int tid  = threadIdx.x;
    const int lane = tid & 31;
    const int w    = tid >> 5;
    const int wM   = (w >> 2) * 32;       // 0 or 32
    const int wN   = (w & 3) * 32;        // 0,32,64,96
    const int m0   = blockIdx.y * 64;
    const int n0   = blockIdx.x * 128;
    const int l    = lane & 15;
    const int aklo = (lane < 16) ? 0 : 8;   // A: lanes<16 hold K{0..7,16..23}
    const int bklo = (lane < 16) ? 0 : 8;   // B: lanes<16 hold K{0..7,16..23}

    // per-thread global staging coordinates
    const int ar  = tid >> 2;             // A row 0..63
    const int acg = (tid & 3) * 8;        // A col group
    const int br0 = tid >> 4;             // B k-row 0..15
    const int br1 = br0 + 16;             // B k-row 16..31
    const int bc  = (tid & 15) * 8;       // B n-col group

    v8f zero = {0.f, 0.f, 0.f, 0.f, 0.f, 0.f, 0.f, 0.f};
    v8f acc[2][2] = {{zero, zero}, {zero, zero}};

    // ---- prologue: stage tile 0 into buffer 0 ----
    float4 aR  = *(const float4*)&A[(size_t)(m0 + ar) * K + acg];
    float4 bR0 = *(const float4*)&Bm[(size_t)br0 * N + n0 + bc];
    float4 bR1 = *(const float4*)&Bm[(size_t)br1 * N + n0 + bc];
    stage_tile(lA[0], lB[0], aR, bR0, bR1, ar, acg, br0, br1, bc);
    __syncthreads();

    int buf = 0;
    for (int k0 = 0; k0 < K; k0 += 32) {
        const bool hasNext = (k0 + 32) < K;
        if (hasNext) {
            const int kn = k0 + 32;
            aR  = *(const float4*)&A[(size_t)(m0 + ar) * K + kn + acg];
            bR0 = *(const float4*)&Bm[(size_t)(kn + br0) * N + n0 + bc];
            bR1 = *(const float4*)&Bm[(size_t)(kn + br1) * N + n0 + bc];
        }
        if (k0 + 64 < K) {   // two tiles ahead -> global_prefetch_b8
            const int kp = k0 + 64;
            __builtin_prefetch(&A[(size_t)(m0 + ar) * K + kp + acg], 0, 1);
            __builtin_prefetch(&Bm[(size_t)(kp + br0) * N + n0 + bc], 0, 1);
            __builtin_prefetch(&Bm[(size_t)(kp + br1) * N + n0 + bc], 0, 1);
        }

        // ---- fragments from current buffer ----
        const _Float16* sA = lA[buf];
        const _Float16* sB = lB[buf];
        HV af[2], bf[2];
#pragma unroll
        for (int mt = 0; mt < 2; ++mt) {
            const _Float16* p = &sA[(wM + mt * 16 + l) * LDA + aklo];
            af[mt].f[0] = *(const float4*)(p);        // K klo..klo+7
            af[mt].f[1] = *(const float4*)(p + 16);   // K klo+16..klo+23
        }
#pragma unroll
        for (int nt = 0; nt < 2; ++nt) {
            const _Float16* p = &sB[(wN + nt * 16 + l) * LDB + bklo];
            bf[nt].f[0] = *(const float4*)(p);        // K bklo..bklo+7
            bf[nt].f[1] = *(const float4*)(p + 16);   // K bklo+16..bklo+23
        }
#pragma unroll
        for (int mt = 0; mt < 2; ++mt)
#pragma unroll
            for (int nt = 0; nt < 2; ++nt)
                acc[mt][nt] = __builtin_amdgcn_wmma_f32_16x16x32_f16(
                    false, af[mt].v, false, bf[nt].v,
                    (short)0, acc[mt][nt], false, false);

        // ---- stage next tile into alternate buffer ----
        if (hasNext)
            stage_tile(lA[buf ^ 1], lB[buf ^ 1], aR, bR0, bR1,
                       ar, acg, br0, br1, bc);
        __syncthreads();
        buf ^= 1;
    }

    // Store: C layout -> VGPR j: lanes0-15 row j, lanes16-31 row j+8; N = lane&15
    const int rsel = (lane < 16) ? 0 : 8;
#pragma unroll
    for (int mt = 0; mt < 2; ++mt) {
#pragma unroll
        for (int nt = 0; nt < 2; ++nt) {
            const int coln = n0 + wN + nt * 16 + l;
            const float bb = bias ? bias[coln] : 0.f;
#pragma unroll
            for (int j = 0; j < 8; ++j) {
                const int row = m0 + wM + mt * 16 + rsel + j;
                float v = acc[mt][nt][j] + bb;
                if (epi == 1) v = gelu_exact(v);
                const size_t o = (size_t)row * N + coln;
                if (out32) out32[o] = v;
                if (out16) out16[o] = (_Float16)v;
            }
        }
    }
}

// ---------------------------------------------------------------------------
// Row LayerNorm (width 768) with optional residual input, optional GELU,
// optional fp32 and f16 outputs. Safe for in-place (res == out32).
// ---------------------------------------------------------------------------
__global__ __launch_bounds__(256) void ln_kernel(
    const float* __restrict__ a, const float* __restrict__ res,
    const float* __restrict__ g, const float* __restrict__ bt,
    int do_gelu, float* __restrict__ out32, _Float16* __restrict__ out16)
{
    const int row = blockIdx.x;
    const int tid = threadIdx.x;
    __shared__ float r1[256], r2[256];

    float vals[3];
    float s1 = 0.f, s2 = 0.f;
#pragma unroll
    for (int j = 0; j < 3; ++j) {
        const int c = tid + j * 256;
        float v = a[(size_t)row * DD + c];
        if (res) v += res[(size_t)row * DD + c];
        vals[j] = v;
        s1 += v;
        s2 += v * v;
    }
    r1[tid] = s1; r2[tid] = s2;
    __syncthreads();
    for (int s = 128; s > 0; s >>= 1) {
        if (tid < s) { r1[tid] += r1[tid + s]; r2[tid] += r2[tid + s]; }
        __syncthreads();
    }
    const float mean = r1[0] * (1.f / DD);
    const float var  = r2[0] * (1.f / DD) - mean * mean;
    const float rstd = rsqrtf(var + 1e-5f);
#pragma unroll
    for (int j = 0; j < 3; ++j) {
        const int c = tid + j * 256;
        float v = (vals[j] - mean) * rstd * g[c] + bt[c];
        if (do_gelu) v = gelu_exact(v);
        const size_t o = (size_t)row * DD + c;
        if (out32) out32[o] = v;
        if (out16) out16[o] = (_Float16)v;
    }
}

// ---------------------------------------------------------------------------
// gate = sigmoid(dot(g1_row[384], w2) + b2)   (g1 already GELU'd by GEMM)
// ---------------------------------------------------------------------------
__global__ __launch_bounds__(128) void gate2_kernel(
    const _Float16* __restrict__ g1, const float* __restrict__ w2,
    const float* __restrict__ b2, float* __restrict__ gate)
{
    const int row = blockIdx.x;
    const int tid = threadIdx.x;
    __shared__ float red[128];
    float s = 0.f;
#pragma unroll
    for (int j = 0; j < 3; ++j) {
        const int c = tid + j * 128;
        s += (float)g1[(size_t)row * 384 + c] * w2[c];
    }
    red[tid] = s;
    __syncthreads();
    for (int st = 64; st > 0; st >>= 1) {
        if (tid < st) red[tid] += red[tid + st];
        __syncthreads();
    }
    if (tid == 0) gate[row] = 1.f / (1.f + expf(-(red[0] + b2[0])));
}

// ---------------------------------------------------------------------------
// gwt = wt * (1 + 0.5*wgate), re-laid out into 2x2-merged regions [B*64, 3072]
// ---------------------------------------------------------------------------
__global__ __launch_bounds__(256) void merge_kernel(
    const _Float16* __restrict__ wt, const float* __restrict__ wgate,
    _Float16* __restrict__ merged)
{
    const int bw  = blockIdx.x;          // b*256 + win
    const int b   = bw >> 8, win = bw & 255;
    const int wy  = win >> 4, wx = win & 15;
    const int rr  = b * RTOK + (wy >> 1) * 8 + (wx >> 1);
    const int off = ((wy & 1) * 2 + (wx & 1)) * DD;
    const float sc = 1.f + 0.5f * wgate[bw];
#pragma unroll
    for (int j = 0; j < 3; ++j) {
        const int c = threadIdx.x + j * 256;
        merged[(size_t)rr * 3072 + off + c] =
            (_Float16)((float)wt[(size_t)bw * DD + c] * sc);
    }
}

// ---------------------------------------------------------------------------
// x = mr * (1 + 0.5*rgate)  -> fp32 + f16
// ---------------------------------------------------------------------------
__global__ __launch_bounds__(256) void scale_kernel(
    const float* __restrict__ mr, const float* __restrict__ rgate,
    float* __restrict__ x32, _Float16* __restrict__ x16)
{
    const int row = blockIdx.x;
    const float sc = 1.f + 0.5f * rgate[row];
#pragma unroll
    for (int j = 0; j < 3; ++j) {
        const int c = threadIdx.x + j * 256;
        const float v = mr[(size_t)row * DD + c] * sc;
        x32[(size_t)row * DD + c] = v;
        x16[(size_t)row * DD + c] = (_Float16)v;
    }
}

// ---------------------------------------------------------------------------
// Multi-head attention over 64 region tokens. One block per (batch, head).
// qkv: [B*64, 2304] fp32. Output o_h: [B*64, 768] f16 (heads re-concatenated).
// ---------------------------------------------------------------------------
__global__ __launch_bounds__(256) void attn_kernel(
    const float* __restrict__ qkv, _Float16* __restrict__ o_h)
{
    const int b = blockIdx.x >> 3;
    const int h = blockIdx.x & 7;
    const int tid = threadIdx.x;

    __shared__ float lkv[RTOK * DH];   // K, later V (24 KB)
    __shared__ float ls[RTOK * RTOK];  // scores / probs (16 KB)

    // load K
    for (int i = tid; i < RTOK * DH; i += 256) {
        const int r = i / DH, d = i % DH;
        lkv[i] = qkv[(size_t)(b * RTOK + r) * 2304 + DD + h * DH + d];
    }
    __syncthreads();

    // scores: thread handles row rq = tid/4, cols ro+4*i
    const int rq = tid >> 2;
    const int ro = tid & 3;
    float acc[16];
#pragma unroll
    for (int i = 0; i < 16; ++i) acc[i] = 0.f;
    for (int d = 0; d < DH; ++d) {
        const float qv = qkv[(size_t)(b * RTOK + rq) * 2304 + h * DH + d];
#pragma unroll
        for (int i = 0; i < 16; ++i) acc[i] += qv * lkv[(ro + 4 * i) * DH + d];
    }
    const float sc = 0.1020620726159658f;   // 1/sqrt(96)
#pragma unroll
    for (int i = 0; i < 16; ++i) ls[rq * RTOK + ro + 4 * i] = acc[i] * sc;
    __syncthreads();

    // softmax per row
    if (tid < RTOK) {
        float m = -3.4e38f;
        for (int j = 0; j < RTOK; ++j) m = fmaxf(m, ls[tid * RTOK + j]);
        float s = 0.f;
        for (int j = 0; j < RTOK; ++j) {
            const float e = expf(ls[tid * RTOK + j] - m);
            ls[tid * RTOK + j] = e;
            s += e;
        }
        const float inv = 1.f / s;
        for (int j = 0; j < RTOK; ++j) ls[tid * RTOK + j] *= inv;
    }
    __syncthreads();

    // load V (reuse lkv)
    for (int i = tid; i < RTOK * DH; i += 256) {
        const int r = i / DH, d = i % DH;
        lkv[i] = qkv[(size_t)(b * RTOK + r) * 2304 + 2 * DD + h * DH + d];
    }
    __syncthreads();

    // O = P @ V
    for (int i = tid; i < RTOK * DH; i += 256) {
        const int r = i / DH, d = i % DH;
        float s = 0.f;
        for (int rk = 0; rk < RTOK; ++rk) s += ls[r * RTOK + rk] * lkv[rk * DH + d];
        o_h[(size_t)(b * RTOK + r) * DD + h * DH + d] = (_Float16)s;
    }
}

// ---------------------------------------------------------------------------
// Class-query head.  logits[b,c] = sum_r softmax_r(cq[c]·kk[b,r]/sqrt(768)) *
//                                  (vv[b,r]·cl_w)  + cl_b
// One block per batch element.
// ---------------------------------------------------------------------------
__global__ __launch_bounds__(256) void cls_kernel(
    const float* __restrict__ kk, const float* __restrict__ vv,
    const float* __restrict__ cq, const float* __restrict__ clw,
    const float* __restrict__ clb, float* __restrict__ out)
{
    const int b = blockIdx.x;
    const int tid = threadIdx.x;
    __shared__ float u[RTOK];
    __shared__ float s[NCLS * RTOK];
    __shared__ float red[256];

    // u[r] = vv[b,r,:] . cl_w   (4 threads per row)
    {
        const int r = tid >> 2, part = tid & 3;
        float p = 0.f;
        const float* vp = vv + (size_t)(b * RTOK + r) * DD + part * 192;
        const float* wp = clw + part * 192;
        for (int j = 0; j < 192; ++j) p += vp[j] * wp[j];
        red[tid] = p;
    }
    __syncthreads();
    if ((tid & 3) == 0)
        u[tid >> 2] = red[tid] + red[tid + 1] + red[tid + 2] + red[tid + 3];
    __syncthreads();

    // scores: 7*64 dots of length 768
    const float sc = 0.03608439182435161f;  // 1/sqrt(768)
    for (int idx = tid; idx < NCLS * RTOK; idx += 256) {
        const int c = idx / RTOK, r = idx % RTOK;
        float p = 0.f;
        const float* qp = cq + (size_t)c * DD;
        const float* kp = kk + (size_t)(b * RTOK + r) * DD;
        for (int d = 0; d < DD; ++d) p += qp[d] * kp[d];
        s[c * RTOK + r] = p * sc;
    }
    __syncthreads();

    if (tid < NCLS) {
        float m = -3.4e38f;
        for (int r = 0; r < RTOK; ++r) m = fmaxf(m, s[tid * RTOK + r]);
        float den = 0.f;
        for (int r = 0; r < RTOK; ++r) {
            const float e = expf(s[tid * RTOK + r] - m);
            s[tid * RTOK + r] = e;
            den += e;
        }
        const float inv = 1.f / den;
        float logit = 0.f;
        for (int r = 0; r < RTOK; ++r) logit += s[tid * RTOK + r] * inv * u[r];
        out[b * NCLS + tid] = logit + clb[0];
    }
}

// ---------------------------------------------------------------------------
// Host orchestration
// ---------------------------------------------------------------------------
static inline void gemm(hipStream_t st, const _Float16* A, const _Float16* B,
                        const float* bias, float* o32, _Float16* o16,
                        int M, int N, int K, int epi) {
    dim3 g(N / 128, M / 64);
    gemm_f16_wmma_kernel<<<g, 256, 0, st>>>(A, B, bias, o32, o16, M, N, K, epi);
}

static inline void cvt(hipStream_t st, const float* in, _Float16* out, int n) {
    cvt_f32_f16_kernel<<<(n + 255) / 256, 256, 0, st>>>(in, out, n);
}

extern "C" void kernel_launch(void* const* d_in, const int* in_sizes, int n_in,
                              void* d_out, int out_size, void* d_ws, size_t ws_size,
                              hipStream_t stream) {
    (void)in_sizes; (void)n_in; (void)out_size; (void)ws_size;

    const float* h_pixel    = (const float*)d_in[0];
    const float* pixel_gate = (const float*)d_in[1];
    const float* wf_w  = (const float*)d_in[2];
    const float* wf_b  = (const float*)d_in[3];
    const float* wf_g  = (const float*)d_in[4];
    const float* wf_bt = (const float*)d_in[5];
    const float* wg_w1 = (const float*)d_in[6];
    const float* wg_b1 = (const float*)d_in[7];
    const float* wg_w2 = (const float*)d_in[8];
    const float* wg_b2 = (const float*)d_in[9];
    const float* rm_w  = (const float*)d_in[10];
    const float* rm_b  = (const float*)d_in[11];
    const float* rm_g  = (const float*)d_in[12];
    const float* rm_bt = (const float*)d_in[13];
    const float* rg_w1 = (const float*)d_in[14];
    const float* rg_b1 = (const float*)d_in[15];
    const float* rg_w2 = (const float*)d_in[16];
    const float* rg_b2 = (const float*)d_in[17];
    const float* qkv_w = (const float*)d_in[18];
    const float* qkv_b = (const float*)d_in[19];
    const float* out_w = (const float*)d_in[20];
    const float* out_b = (const float*)d_in[21];
    const float* ln1_g = (const float*)d_in[22];
    const float* ln1_b = (const float*)d_in[23];
    const float* ln2_g = (const float*)d_in[24];
    const float* ln2_b = (const float*)d_in[25];
    const float* ff1_w = (const float*)d_in[26];
    const float* ff1_b = (const float*)d_in[27];
    const float* ff2_w = (const float*)d_in[28];
    const float* ff2_b = (const float*)d_in[29];
    const float* cq    = (const float*)d_in[30];
    const float* rk_w  = (const float*)d_in[31];
    const float* rk_b  = (const float*)d_in[32];
    const float* rv_w  = (const float*)d_in[33];
    const float* rv_b  = (const float*)d_in[34];
    const float* cl_w  = (const float*)d_in[35];
    const float* cl_b  = (const float*)d_in[36];

    // --- workspace bump allocator (256B aligned) ---
    char* ws = (char*)d_ws;
    size_t off = 0;
    auto alloc = [&](size_t bytes) -> void* {
        void* p = ws + off;
        off = (off + bytes + 255) & ~(size_t)255;
        return p;
    };

    _Float16* Wf   = (_Float16*)alloc((size_t)1536 * 768 * 2);
    _Float16* Wg1  = (_Float16*)alloc((size_t)768 * 384 * 2);
    _Float16* Rm   = (_Float16*)alloc((size_t)3072 * 768 * 2);
    _Float16* Rg1  = (_Float16*)alloc((size_t)768 * 384 * 2);
    _Float16* Wqkv = (_Float16*)alloc((size_t)768 * 2304 * 2);
    _Float16* Wout = (_Float16*)alloc((size_t)768 * 768 * 2);
    _Float16* Wff1 = (_Float16*)alloc((size_t)768 * 1536 * 2);
    _Float16* Wff2 = (_Float16*)alloc((size_t)1536 * 768 * 2);
    _Float16* Wrk  = (_Float16*)alloc((size_t)768 * 768 * 2);
    _Float16* Wrv  = (_Float16*)alloc((size_t)768 * 768 * 2);

    _Float16* cat_h  = (_Float16*)alloc((size_t)4096 * 1536 * 2);
    float*    y32    = (float*)alloc((size_t)4096 * 768 * 4);   // max raw GEMM out
    _Float16* wt_h   = (_Float16*)alloc((size_t)4096 * 768 * 2);
    _Float16* g1_h   = (_Float16*)alloc((size_t)4096 * 384 * 2);
    float*    wgate  = (float*)alloc((size_t)4096 * 4);
    _Float16* merged = (_Float16*)alloc((size_t)1024 * 3072 * 2);
    float*    mr32   = (float*)alloc((size_t)1024 * 768 * 4);
    _Float16* mr_h   = (_Float16*)alloc((size_t)1024 * 768 * 2);
    float*    rgate  = (float*)alloc((size_t)1024 * 4);
    float*    x32    = (float*)alloc((size_t)1024 * 768 * 4);
    _Float16* x_h    = (_Float16*)alloc((size_t)1024 * 768 * 2);
    _Float16* o_h    = (_Float16*)alloc((size_t)1024 * 768 * 2);
    _Float16* ff1_h  = (_Float16*)alloc((size_t)1024 * 1536 * 2);
    float*    kk32   = (float*)alloc((size_t)1024 * 768 * 4);
    float*    vv32   = (float*)alloc((size_t)1024 * 768 * 4);

    // --- weights fp32 -> f16 ---
    cvt(stream, wf_w,  Wf,   1536 * 768);
    cvt(stream, wg_w1, Wg1,  768 * 384);
    cvt(stream, rm_w,  Rm,   3072 * 768);
    cvt(stream, rg_w1, Rg1,  768 * 384);
    cvt(stream, qkv_w, Wqkv, 768 * 2304);
    cvt(stream, out_w, Wout, 768 * 768);
    cvt(stream, ff1_w, Wff1, 768 * 1536);
    cvt(stream, ff2_w, Wff2, 1536 * 768);
    cvt(stream, rk_w,  Wrk,  768 * 768);
    cvt(stream, rv_w,  Wrv,  768 * 768);

    // --- 1. window pooling (reg + shifted), write [B*256, 1536] f16 ---
    pool_kernel<<<dim3(BD * NWIN, 2), 256, 0, stream>>>(h_pixel, pixel_gate, cat_h);

    // --- 2. window fuse: GEMM -> LN -> GELU ---
    gemm(stream, cat_h, Wf, wf_b, y32, nullptr, 4096, 768, 1536, 0);
    ln_kernel<<<4096, 256, 0, stream>>>(y32, nullptr, wf_g, wf_bt, 1, nullptr, wt_h);

    // --- 3. window gate ---
    gemm(stream, wt_h, Wg1, wg_b1, nullptr, g1_h, 4096, 384, 768, 1);
    gate2_kernel<<<4096, 128, 0, stream>>>(g1_h, wg_w2, wg_b2, wgate);

    // --- 4. gwt + 2x2 region merge ---
    merge_kernel<<<4096, 256, 0, stream>>>(wt_h, wgate, merged);

    // --- 5. region merge proj: GEMM -> LN -> GELU ---
    gemm(stream, merged, Rm, rm_b, y32, nullptr, 1024, 768, 3072, 0);
    ln_kernel<<<1024, 256, 0, stream>>>(y32, nullptr, rm_g, rm_bt, 1, mr32, mr_h);

    // --- 6. region gate, x = mr * (1 + 0.5*rgate) ---
    gemm(stream, mr_h, Rg1, rg_b1, nullptr, g1_h, 1024, 384, 768, 1);
    gate2_kernel<<<1024, 128, 0, stream>>>(g1_h, rg_w2, rg_b2, rgate);
    scale_kernel<<<1024, 256, 0, stream>>>(mr32, rgate, x32, x_h);

    // --- 7. transformer layer (post-norm) ---
    gemm(stream, x_h, Wqkv, qkv_b, y32, nullptr, 1024, 2304, 768, 0);
    attn_kernel<<<BD * HEADS, 256, 0, stream>>>(y32, o_h);
    gemm(stream, o_h, Wout, out_b, y32, nullptr, 1024, 768, 768, 0);
    ln_kernel<<<1024, 256, 0, stream>>>(y32, x32, ln1_g, ln1_b, 0, x32, x_h);
    gemm(stream, x_h, Wff1, ff1_b, nullptr, ff1_h, 1024, 1536, 768, 1);
    gemm(stream, ff1_h, Wff2, ff2_b, y32, nullptr, 1024, 768, 1536, 0);
    ln_kernel<<<1024, 256, 0, stream>>>(y32, x32, ln2_g, ln2_b, 0, x32, x_h);

    // --- 8. class-query head ---
    gemm(stream, x_h, Wrk, rk_b, kk32, nullptr, 1024, 768, 768, 0);
    gemm(stream, x_h, Wrv, rv_b, vv32, nullptr, 1024, 768, 768, 0);
    cls_kernel<<<BD, 256, 0, stream>>>(kk32, vv32, cq, cl_w, cl_b, (float*)d_out);
}
